// SlotAttention_90718299226711
// MI455X (gfx1250) — matrix operand order repaired and checked by reading, not compiled, not run on hardware
//
#include <hip/hip_runtime.h>
#include <math.h>

typedef __attribute__((ext_vector_type(16))) __bf16 v16bf;
typedef __attribute__((ext_vector_type(8)))  float  v8f;
typedef unsigned int u32x4 __attribute__((ext_vector_type(4)));
typedef int          i32x8 __attribute__((ext_vector_type(8)));
typedef int          i32x4 __attribute__((ext_vector_type(4)));

#define B_   32
#define NIN  4096
#define INP  512
#define NS   8
#define D_   256
#define MLPH 512
#define MKV  (B_ * NIN)   // 131072 rows

#define A_ROWS   128
#define A_PITCH  1056     // 1024B row + 8 dwords TDM pad -> bank-conflict-free frags
#define A_LDS_BYTES (A_ROWS * A_PITCH)   // 135168

union Frag16 { uint4 u[2]; v16bf v; };

__device__ __forceinline__ unsigned short f2bf(float f) {
  unsigned u = __builtin_bit_cast(unsigned, f);
  u += 0x7FFFu + ((u >> 16) & 1u);          // round-to-nearest-even
  return (unsigned short)(u >> 16);
}

__device__ __forceinline__ float wave_reduce_add(float v) {
  for (int off = 16; off >= 1; off >>= 1) v += __shfl_xor(v, off, 32);
  return v;
}

// ---------------------------------------------------------------- LN(inputs) -> bf16
__global__ void ln_x_kernel(const float* __restrict__ xin, unsigned short* __restrict__ xbf) {
  int wave = threadIdx.x >> 5, lane = threadIdx.x & 31;
  long row = (long)blockIdx.x * 8 + wave;
  const float* p = xin + row * INP;
  float vals[16];
  float s = 0.f, s2 = 0.f;
  for (int i = 0; i < 16; ++i) { float f = p[lane * 16 + i]; vals[i] = f; s += f; s2 += f * f; }
  s = wave_reduce_add(s); s2 = wave_reduce_add(s2);
  float m = s * (1.0f / INP);
  float var = s2 * (1.0f / INP) - m * m;
  float r = rsqrtf(var + 1e-5f);
  unsigned short* o = xbf + row * INP;
  for (int i = 0; i < 16; ++i) o[lane * 16 + i] = f2bf((vals[i] - m) * r);
}

// ------------------------------------------------- pack [Wk|Wv] into B-fragment layout
// layout: [ct(32)][t(16)][lane(32)][j(16)] ; col = ct*16 + lane%16 ; K = t*32 + (j/8)*16 + (lane/16)*8 + j%8
__global__ void pack_wkv_kernel(const float* __restrict__ Wk, const float* __restrict__ Wv,
                                unsigned short* __restrict__ wp) {
  int id = blockIdx.x * 256 + threadIdx.x;          // 262144
  int j = id & 15, lane = (id >> 4) & 31, t = (id >> 9) & 15, ct = id >> 13;
  int col = ct * 16 + (lane & 15);
  int k = t * 32 + ((j >> 3) << 4) + ((lane >> 4) << 3) + (j & 7);
  float v = (col < 256) ? Wk[k * 256 + col] : Wv[k * 256 + (col - 256)];
  wp[id] = f2bf(v);
}

// ------------------------- k,v = x @ [Wk|Wv] : TDM-staged A tile in LDS + WMMA bf16
__global__ void gemm_kv_kernel(const unsigned short* __restrict__ xbf,
                               const unsigned short* __restrict__ wp,
                               unsigned short* __restrict__ kbf,
                               unsigned short* __restrict__ vbf) {
  extern __shared__ unsigned char lds_a[];   // A_ROWS rows * A_PITCH bytes
  int wave = threadIdx.x >> 5, lane = threadIdx.x & 31, half = lane >> 4;
  long rowbase = (long)blockIdx.x * A_ROWS;

  // ---- Tensor Data Mover: DMA 128x512 bf16 tile (row pitch padded by 8 dwords) ----
  if (wave == 0) {
    unsigned long gaddr = (unsigned long)(const void*)(xbf + rowbase * INP);
    unsigned lds0 = (unsigned)(unsigned long)(void*)&lds_a[0];
    u32x4 g0;
    g0[0] = 1u;                                             // count=1, user mode
    g0[1] = lds0;                                           // lds_addr
    g0[2] = (unsigned)(gaddr & 0xFFFFFFFFu);                // global_addr[31:0]
    g0[3] = (unsigned)((gaddr >> 32) & 0x01FFFFFFu) | 0x80000000u; // addr[56:32] | type=2
    i32x8 g1;
    g1[0] = (1 << 16) | (1 << 20) | (7 << 22) | (7 << 25);  // 2B elems, pad_en, 256dw interval, 8dw pad
    g1[1] = (512 << 16);                                    // tensor_dim0 = 512 (low 16)
    g1[2] = (128 << 16);                                    // tensor_dim0 hi=0 | tensor_dim1 = 128
    g1[3] = (512 << 16);                                    // tensor_dim1 hi=0 | tile_dim0 = 512
    g1[4] = 128;                                            // tile_dim1 = 128, tile_dim2 = 0
    g1[5] = 512;                                            // tensor_dim0_stride = 512
    g1[6] = 0;
    g1[7] = 0;
    i32x4 z4 = {0, 0, 0, 0};
    i32x8 z8 = {0, 0, 0, 0, 0, 0, 0, 0};
    __builtin_amdgcn_tensor_load_to_lds(g0, g1, z4, z4, z8, 0);
    __builtin_amdgcn_s_wait_tensorcnt(0);
  }
  __syncthreads();

  const unsigned char* arow = lds_a + (wave * 16 + (lane & 15)) * A_PITCH;
  for (int yc = 0; yc < 8; ++yc) {               // all 512 output cols per block
    // stop LICM from hoisting 16 A-fragments (=> 128 VGPRs + scratch spills):
    // force the ds_load_b128 reads to be re-issued inside each yc pass.
    asm volatile("" ::: "memory");
    v8f acc[4] = {};
    for (int t = 0; t < 16; ++t) {
      int k0 = t * 32;
      Frag16 a;
      a.u[0] = *(const uint4*)(arow + (k0 + half * 8) * 2);
      a.u[1] = *(const uint4*)(arow + (k0 + half * 8 + 16) * 2);
      for (int ct = 0; ct < 4; ++ct) {
        int ctg = yc * 4 + ct;
        Frag16 b;
        b.v = *(const v16bf*)(wp + (((long)ctg * 16 + t) * 32 + lane) * 16);
        acc[ct] = __builtin_amdgcn_wmma_f32_16x16x32_bf16(false, a.v, false, b.v,
                                                          (short)0, acc[ct], false, false);
      }
    }
    for (int ct = 0; ct < 4; ++ct) {
      int colg = yc * 64 + ct * 16 + (lane & 15);
      for (int i = 0; i < 8; ++i) {
        long r = rowbase + wave * 16 + i + half * 8;
        unsigned short bv = f2bf(acc[ct][i]);
        if (colg < 256) kbf[r * D_ + colg] = bv;
        else            vbf[r * D_ + (colg - 256)] = bv;
      }
    }
  }
}

// ---------------------------------------- repack v into B-fragment layout (done once)
// layout: [b][ct(16)][t(128)][lane(32)][j(16)] ; col = ct*16+lane%16 ; n = t*32+(j/8)*16+(lane/16)*8+j%8
__global__ void pack_v_kernel(const unsigned short* __restrict__ vbf,
                              unsigned short* __restrict__ vp) {
  long id = (long)blockIdx.x * 256 + threadIdx.x;   // 33554432
  int j = (int)(id & 15), lane = (int)((id >> 4) & 31);
  int t = (int)((id >> 9) & 127), ct = (int)((id >> 16) & 15), b = (int)(id >> 20);
  int col = ct * 16 + (lane & 15);
  int n = t * 32 + ((j >> 3) << 4) + ((lane >> 4) << 3) + (j & 7);
  vp[id] = vbf[((long)(b * NIN + n)) * D_ + col];
}

// ---------------------------------------------- per-iteration: s = LN(slots), q = s@Wq
__global__ void slots_prep_kernel(const float* __restrict__ slots, const float* __restrict__ Wq,
                                  float* __restrict__ snorm, float* __restrict__ q) {
  __shared__ float sv[D_], r1[D_], r2[D_];
  int row = blockIdx.x, t = threadIdx.x;
  float x = slots[row * D_ + t];
  r1[t] = x; r2[t] = x * x;
  __syncthreads();
  for (int off = 128; off >= 1; off >>= 1) {
    if (t < off) { r1[t] += r1[t + off]; r2[t] += r2[t + off]; }
    __syncthreads();
  }
  float m = r1[0] * (1.0f / D_);
  float var = r2[0] * (1.0f / D_) - m * m;
  float rs = rsqrtf(var + 1e-5f);
  float y = (x - m) * rs;
  sv[t] = y; snorm[row * D_ + t] = y;
  __syncthreads();
  float acc = 0.f;
  for (int k = 0; k < D_; ++k) acc += sv[k] * Wq[k * D_ + t];
  q[row * D_ + t] = acc;
}

// ------------------------------- pack q into B-fragments (slots padded 8->16 with zeros)
__global__ void pack_q_kernel(const float* __restrict__ q, unsigned short* __restrict__ qp,
                              float* __restrict__ colsum) {
  int id = blockIdx.x * 256 + threadIdx.x;          // 131072
  int j = id & 15, lane = (id >> 4) & 31, t = (id >> 9) & 7, b = id >> 12;
  int slot = lane & 15;
  int k = t * 32 + ((j >> 3) << 4) + ((lane >> 4) << 3) + (j & 7);
  float v = (slot < NS) ? q[(b * NS + slot) * D_ + k] : 0.f;
  qp[id] = f2bf(v);
  if (id < B_ * NS) colsum[id] = 0.f;
}

// ----------------------------------------------- logits = k . q^T * scale  (WMMA bf16)
__global__ void logits_gemm_kernel(const unsigned short* __restrict__ kbf,
                                   const unsigned short* __restrict__ qp,
                                   float* __restrict__ logits) {
  int wave = threadIdx.x >> 5, lane = threadIdx.x & 31, half = lane >> 4;
  int b = blockIdx.x >> 5;
  int mt = blockIdx.x & 31;
  int rowbase = mt * 128 + wave * 16;
  const unsigned short* ap = kbf + ((long)(b * NIN) + rowbase + (lane & 15)) * D_;
  v8f acc = {};
  for (int t = 0; t < 8; ++t) {
    int k0 = t * 32;
    Frag16 a;
    a.u[0] = *(const uint4*)(ap + k0 + half * 8);
    a.u[1] = *(const uint4*)(ap + k0 + half * 8 + 16);
    if (t < 7) __builtin_prefetch(ap + k0 + 32, 0, 1);
    Frag16 bfr;
    bfr.v = *(const v16bf*)(qp + (((long)b * 8 + t) * 32 + lane) * 16);
    acc = __builtin_amdgcn_wmma_f32_16x16x32_bf16(false, a.v, false, bfr.v,
                                                  (short)0, acc, false, false);
  }
  int slot = lane & 15;
  if (slot < NS) {
    for (int i = 0; i < 8; ++i) {
      int n = rowbase + i + half * 8;
      logits[((long)(b * NIN) + n) * NS + slot] = acc[i] * 0.0625f;  // 1/sqrt(256)
    }
  }
}

// -------- softmax over slots + eps; accumulate column sums; emit attn as A-fragments
// A layout: [b][t(128)][lane(32)][j(16)] ; lane<16: row=lane, K runs [k0,k0+8)+[k0+16,k0+24)
__global__ void softmax_attn_kernel(const float* __restrict__ logits,
                                    unsigned short* __restrict__ apk,
                                    float* __restrict__ colsum) {
  int id = blockIdx.x * 256 + threadIdx.x;          // B*NIN
  int b = id >> 12, n = id & 4095;
  const float* lp = logits + (long)id * NS;
  float l[8], mx = -1e30f;
  for (int i = 0; i < 8; ++i) { l[i] = lp[i]; mx = fmaxf(mx, l[i]); }
  float sum = 0.f;
  for (int i = 0; i < 8; ++i) { l[i] = __expf(l[i] - mx); sum += l[i]; }
  float inv = 1.0f / sum;
  float a8[8];
  for (int i = 0; i < 8; ++i) {
    a8[i] = l[i] * inv + 1e-8f;
    atomicAdd(&colsum[b * NS + i], a8[i]);
  }
  int t = n >> 5, rel = n & 31;
  int halfsel = (rel >> 3) & 1;
  int j = ((rel & 16) ? 8 : 0) + (rel & 7);
  unsigned short* base = apk + (((long)b * 128 + t) * 32) * 16;
  for (int s16 = 0; s16 < 16; ++s16) {
    int lane = halfsel * 16 + s16;
    base[lane * 16 + j] = (s16 < NS) ? f2bf(a8[s16]) : (unsigned short)0;
  }
}

// ------------------------------- updates = (attn^T @ v) / colsum  (WMMA bf16, K=4096)
__global__ void updates_gemm_kernel(const unsigned short* __restrict__ apk,
                                    const unsigned short* __restrict__ vp,
                                    const float* __restrict__ colsum,
                                    float* __restrict__ upd) {
  int wave = threadIdx.x >> 5, lane = threadIdx.x & 31, half = lane >> 4;
  int b = blockIdx.x;
  v8f acc[2] = {};
  for (int t = 0; t < 128; ++t) {
    Frag16 a;
    a.v = *(const v16bf*)(apk + (((long)b * 128 + t) * 32 + lane) * 16);
    for (int c = 0; c < 2; ++c) {
      int ct = wave * 2 + c;
      Frag16 bf;
      bf.v = *(const v16bf*)(vp + ((((long)b * 16 + ct) * 128 + t) * 32 + lane) * 16);
      acc[c] = __builtin_amdgcn_wmma_f32_16x16x32_bf16(false, a.v, false, bf.v,
                                                       (short)0, acc[c], false, false);
    }
  }
  for (int c = 0; c < 2; ++c) {
    int col = (wave * 2 + c) * 16 + (lane & 15);
    for (int i = 0; i < 8; ++i) {
      int s = i + half * 8;
      if (s < NS) upd[(b * NS + s) * D_ + col] = acc[c][i] / colsum[b * NS + s];
    }
  }
}

// ------------------------------------------------------------- GRU cell (fp32 scalar)
__global__ void gru_kernel(const float* __restrict__ upd, const float* __restrict__ snorm,
                           const float* __restrict__ wih, const float* __restrict__ whh,
                           const float* __restrict__ bih, const float* __restrict__ bhh,
                           float* __restrict__ slots) {
  __shared__ float xu[D_], hh[D_];
  int row = blockIdx.x, t = threadIdx.x;
  xu[t] = upd[row * D_ + t];
  hh[t] = snorm[row * D_ + t];
  __syncthreads();
  float gx[3], gh[3];
  for (int g = 0; g < 3; ++g) {
    int o = g * D_ + t;
    float ax = bih[o], ah = bhh[o];
    const float* wi = wih + (long)o * D_;
    const float* wh = whh + (long)o * D_;
    for (int k = 0; k < D_; ++k) { ax += xu[k] * wi[k]; ah += hh[k] * wh[k]; }
    gx[g] = ax; gh[g] = ah;
  }
  float r = 1.f / (1.f + __expf(-(gx[0] + gh[0])));
  float z = 1.f / (1.f + __expf(-(gx[1] + gh[1])));
  float nn = tanhf(gx[2] + r * gh[2]);
  slots[row * D_ + t] = (1.f - z) * nn + z * hh[t];
}

// ---------------------------------------------- MLP residual: slots += W2 relu(W1 LN(slots))
__global__ void mlp_kernel(const float* __restrict__ w1, const float* __restrict__ b1,
                           const float* __restrict__ w2, const float* __restrict__ b2,
                           float* __restrict__ slots) {
  __shared__ float sx[D_], y[D_], hid[MLPH], r1[D_], r2[D_];
  int row = blockIdx.x, t = threadIdx.x;
  float x = slots[row * D_ + t];
  sx[t] = x; r1[t] = x; r2[t] = x * x;
  __syncthreads();
  for (int off = 128; off >= 1; off >>= 1) {
    if (t < off) { r1[t] += r1[t + off]; r2[t] += r2[t + off]; }
    __syncthreads();
  }
  float m = r1[0] * (1.0f / D_);
  float var = r2[0] * (1.0f / D_) - m * m;
  float rs = rsqrtf(var + 1e-5f);
  y[t] = (x - m) * rs;
  __syncthreads();
  for (int rep = 0; rep < 2; ++rep) {
    int j = rep * D_ + t;
    float a = b1[j];
    for (int k = 0; k < D_; ++k) a += y[k] * w1[(long)k * MLPH + j];
    hid[j] = fmaxf(a, 0.f);
  }
  __syncthreads();
  float o = b2[t];
  for (int j = 0; j < MLPH; ++j) o += hid[j] * w2[(long)j * D_ + t];
  slots[row * D_ + t] = sx[t] + o;
}

// ----------------------------------------------- slots init: mu + exp(log_sigma)*noise
__global__ void init_slots_kernel(const float* __restrict__ noise, const float* __restrict__ mu,
                                  const float* __restrict__ ls, float* __restrict__ slots) {
  int id = blockIdx.x * 256 + threadIdx.x;          // 65536
  int d = id & (D_ - 1);
  slots[id] = mu[d] + __expf(ls[d]) * noise[id];
}

extern "C" void kernel_launch(void* const* d_in, const int* in_sizes, int n_in,
                              void* d_out, int out_size, void* d_ws, size_t ws_size,
                              hipStream_t stream) {
  const float* inputs = (const float*)d_in[0];
  const float* noise  = (const float*)d_in[1];
  const float* Wk     = (const float*)d_in[2];
  const float* Wv     = (const float*)d_in[3];
  const float* Wq     = (const float*)d_in[4];
  const float* mu     = (const float*)d_in[5];
  const float* ls     = (const float*)d_in[6];
  const float* wih    = (const float*)d_in[7];
  const float* whh    = (const float*)d_in[8];
  const float* bih    = (const float*)d_in[9];
  const float* bhh    = (const float*)d_in[10];
  const float* w1     = (const float*)d_in[11];
  const float* b1     = (const float*)d_in[12];
  const float* w2     = (const float*)d_in[13];
  const float* b2     = (const float*)d_in[14];
  float* slots = (float*)d_out;   // slots live in d_out; final state is the answer

  char* ws = (char*)d_ws;
  size_t off = 0;
  auto alloc = [&](size_t bytes) -> void* {
    void* p = ws + off;
    off = (off + bytes + 255) & ~(size_t)255;
    return p;
  };
  unsigned short* xbf  = (unsigned short*)alloc((size_t)MKV * INP * 2);      // 134 MB
  unsigned short* kbf  = (unsigned short*)alloc((size_t)MKV * D_ * 2);       //  67 MB
  unsigned short* vbf  = (unsigned short*)alloc((size_t)MKV * D_ * 2);       //  67 MB
  unsigned short* vp   = (unsigned short*)alloc((size_t)MKV * D_ * 2);       //  67 MB
  unsigned short* wp   = (unsigned short*)alloc((size_t)INP * 512 * 2);
  float* snorm         = (float*)alloc((size_t)B_ * NS * D_ * 4);
  float* q             = (float*)alloc((size_t)B_ * NS * D_ * 4);
  unsigned short* qp   = (unsigned short*)alloc((size_t)B_ * 8 * 32 * 16 * 2);
  unsigned short* apk  = (unsigned short*)alloc((size_t)B_ * 128 * 32 * 16 * 2);
  float* logits        = (float*)alloc((size_t)B_ * NIN * NS * 4);
  float* colsum        = (float*)alloc((size_t)B_ * NS * 4);
  float* upd           = (float*)alloc((size_t)B_ * NS * D_ * 4);

  init_slots_kernel<<<(B_ * NS * D_) / 256, 256, 0, stream>>>(noise, mu, ls, slots);
  ln_x_kernel<<<MKV / 8, 256, 0, stream>>>(inputs, xbf);
  pack_wkv_kernel<<<262144 / 256, 256, 0, stream>>>(Wk, Wv, wp);
  gemm_kv_kernel<<<MKV / A_ROWS, 256, A_LDS_BYTES, stream>>>(xbf, wp, kbf, vbf);
  pack_v_kernel<<<(MKV * D_) / 256, 256, 0, stream>>>(vbf, vp);

  for (int it = 0; it < 3; ++it) {
    slots_prep_kernel<<<B_ * NS, D_, 0, stream>>>(slots, Wq, snorm, q);
    pack_q_kernel<<<131072 / 256, 256, 0, stream>>>(q, qp, colsum);
    logits_gemm_kernel<<<MKV / 128, 256, 0, stream>>>(kbf, qp, logits);
    softmax_attn_kernel<<<MKV / 256, 256, 0, stream>>>(logits, apk, colsum);
    updates_gemm_kernel<<<B_, 256, 0, stream>>>(apk, vp, colsum, upd);
    gru_kernel<<<B_ * NS, D_, 0, stream>>>(upd, snorm, wih, whh, bih, bhh, slots);
    mlp_kernel<<<B_ * NS, D_, 0, stream>>>(w1, b1, w2, b2, slots);
  }
}
